// ChannelAttention_89936615178480
// MI455X (gfx1250) — matrix operand order repaired and checked by reading, not compiled, not run on hardware
//
#include <hip/hip_runtime.h>
#include <hip/hip_bf16.h>
#include <math.h>

// ---------------------------------------------------------------------------
// Problem sizes (fixed by the reference)
// ---------------------------------------------------------------------------
#define BB   128          // batch
#define LL   196          // locations (K of the big GEMM)
#define CC   2048         // channels (M of the big GEMM)
#define HH   512          // att_hid_size (N of the big GEMM)
#define DD   1024         // decoder hidden
#define LP   232          // LDS row stride in halves (16B-aligned, conflict pad)
#define NKT  7            // K tiles of 32 (K padded 196 -> 224)
#define NHT  32           // H chunks of 16
#define CTILE 128         // channels per block (8 waves x 16 rows)
#define BSTAGE_BYTES 7168 // one H-chunk of packed B: 7*32*16 halves = 7168 B
#define BSTAGE_CHUNKS 448 // BSTAGE_BYTES / 16

typedef __attribute__((ext_vector_type(16))) __bf16 v16bf;
typedef __attribute__((ext_vector_type(8)))  __bf16 v8bf;
typedef __attribute__((ext_vector_type(8)))  float  v8f;

__device__ __forceinline__ unsigned short f2bf(float f) {
    unsigned u = __float_as_uint(f);
    u += 0x7FFFu + ((u >> 16) & 1u);        // round to nearest even
    return (unsigned short)(u >> 16);
}

// Branch-free tanh: prefer the gfx1250 V_TANH_F32 transcendental.
__device__ __forceinline__ float dev_tanh(float x) {
#if __has_builtin(__builtin_amdgcn_tanhf)
    return __builtin_amdgcn_tanhf(x);
#else
    float a = __builtin_fabsf(x);
    float e = __expf(-2.0f * a);                       // v_exp_f32, e in (0,1]
    float t = (1.0f - e) * __frcp_rn(1.0f + e);
    return __builtin_copysignf(t, x);
#endif
}

// CDNA5 async copy: global -> LDS, tracked with ASYNCcnt.
// lds_off is the raw LDS byte offset (the single __shared__ array sits at 0).
__device__ __forceinline__ void async_b128(unsigned lds_off, const void* g) {
    asm volatile("global_load_async_to_lds_b128 %0, %1, off"
                 :: "v"(lds_off), "v"((unsigned long long)g)
                 : "memory");
}
__device__ __forceinline__ void wait_async0() {
    asm volatile("s_wait_asynccnt 0" ::: "memory");
}

// ---------------------------------------------------------------------------
// Kernel 1: pack Wc (196x512 fp32) into bf16 WMMA B-fragments.
// Fragment (hh, kt): 32 lanes x 16 halves, lane-major.
//   lane < 16 : column n = lane,    K = kt*32 + 0..15
//   lane >= 16: column n = lane-16, K = kt*32 + 16..31
// ---------------------------------------------------------------------------
__global__ void k_pack_wc(const float* __restrict__ Wc, unsigned short* __restrict__ wcp) {
    int idx = blockIdx.x * 256 + threadIdx.x;       // 32*7*32*16 = 114688
    if (idx >= NHT * NKT * 32 * 16) return;
    int j    = idx & 15;
    int lane = (idx >> 4) & 31;
    int t    = idx >> 9;
    int kt   = t % NKT;
    int hh   = t / NKT;
    int k    = kt * 32 + ((lane >> 4) & 1) * 16 + j;
    int hcol = hh * 16 + (lane & 15);
    float v  = (k < LL) ? Wc[k * HH + hcol] : 0.0f;
    wcp[idx] = f2bf(v);
}

// ---------------------------------------------------------------------------
// Kernel 2: ph[b,h] = h_vec . Wh[:,h] + bh[h] + bc[h]   (bc folded in)
// ---------------------------------------------------------------------------
__global__ void k_ph(const float* __restrict__ h, const float* __restrict__ Wh,
                     const float* __restrict__ bh, const float* __restrict__ bc,
                     float* __restrict__ ph) {
    int idx = blockIdx.x * 256 + threadIdx.x;       // 128*512
    int b = idx >> 9, j = idx & 511;
    const float* hp = h + b * DD;
    float s = 0.f;
    #pragma unroll 4
    for (int k = 0; k < DD; ++k) s = fmaf(hp[k], Wh[k * HH + j], s);
    ph[idx] = s + bh[j] + bc[j];
}

// ---------------------------------------------------------------------------
// Kernel 3: fused scores[b,c] = wa . tanh( (x_b @ Wc)[c,:] + ph[b,:] ) + ba
// Block = (c-tile of 128, batch). 8 waves, wave owns 16 channel rows.
// LDS reuse timeline (single 59392 B array, LDS offset 0):
//   phase 1: whole array = A-tile (bf16, transposed, K-padded)
//   phase 2: [0,14336) = B ping-pong (async-filled), [16384,24576) = reduce
// ---------------------------------------------------------------------------
__global__ void __launch_bounds__(256)
k_scores(const float* __restrict__ att, const float* __restrict__ ph,
         const float* __restrict__ wa, const unsigned short* __restrict__ wcp,
         const float* __restrict__ ba, float* __restrict__ scores) {
    __shared__ unsigned short tile[CTILE * LP];          // 59392 B, at LDS offset 0
    float* red = (float*)((char*)tile + 16384);          // 8 KB overlay (past B bufs)

    const int b   = blockIdx.y;
    const int c0  = blockIdx.x * CTILE;
    const int tid = threadIdx.x;
    const int lane = tid & 31;
    const int wv   = tid >> 5;
    const int m    = lane & 15;                          // A-row / D-column in wave
    const int hi16 = (lane >> 4) & 1;

    // ---- phase 1: stage att[b, l, c0..c0+127] transposed+converted to LDS ----
    const float* ab = att + (size_t)b * LL * CC + c0;
    for (int i = tid; i < LP * (CTILE / 4); i += 256) {  // 232*32 iters of float4
        int c4 = (i & 31) * 4;
        int l  = i >> 5;                                 // 0..231
        float4 v = make_float4(0.f, 0.f, 0.f, 0.f);
        if (l < LL) {
            v = *(const float4*)(ab + (size_t)l * CC + c4);
            if (l + 8 < LL) __builtin_prefetch(ab + (size_t)(l + 8) * CC + c4, 0, 0);
        }
        tile[(c4 + 0) * LP + l] = f2bf(v.x);
        tile[(c4 + 1) * LP + l] = f2bf(v.y);
        tile[(c4 + 2) * LP + l] = f2bf(v.z);
        tile[(c4 + 3) * LP + l] = f2bf(v.w);
    }
    __syncthreads();

    // ---- pull this wave's A fragments into registers (resident for all H) ----
    const unsigned short* rowp = &tile[(wv * 16 + m) * LP];
    v16bf afrag[NKT];
    #pragma unroll
    for (int kt = 0; kt < NKT; ++kt) {
        v8bf lo = *(const v8bf*)(rowp + kt * 32 + hi16 * 16);       // K base+0..7
        v8bf hi = *(const v8bf*)(rowp + kt * 32 + hi16 * 16 + 8);   // K base+8..15
        afrag[kt] = __builtin_shufflevector(lo, hi,
            0, 1, 2, 3, 4, 5, 6, 7, 8, 9, 10, 11, 12, 13, 14, 15);
    }
    __syncthreads();   // A-tile region is now dead; safe to overlay B/red

    // ---- phase 2: sweep H; B fragments double-buffered through LDS via
    //      global_load_async_to_lds_b128 (ASYNCcnt), shared by all 8 waves ----
    float part[8];
    #pragma unroll
    for (int i = 0; i < 8; ++i) part[i] = 0.f;
    const float* phb = ph + b * HH;

    // prologue: issue stage 0 into buffer 0
    {
        const char* src = (const char*)wcp;
        async_b128((unsigned)(tid * 16), src + (size_t)tid * 16);
        int j2 = tid + 256;
        if (j2 < BSTAGE_CHUNKS) async_b128((unsigned)(j2 * 16), src + (size_t)j2 * 16);
    }

    for (int s = 0; s < NHT; ++s) {
        wait_async0();          // this wave's loads for stage s are in LDS
        __syncthreads();        // all waves: stage s visible; stage s-1 buffer free

        if (s + 1 < NHT) {      // overlap: issue stage s+1 into the other buffer
            const char* src = (const char*)wcp + (size_t)(s + 1) * BSTAGE_BYTES;
            unsigned base = (unsigned)(((s + 1) & 1) * BSTAGE_BYTES);
            async_b128(base + tid * 16, src + (size_t)tid * 16);
            int j2 = tid + 256;
            if (j2 < BSTAGE_CHUNKS) async_b128(base + j2 * 16, src + (size_t)j2 * 16);
        }

        const unsigned short* lb = tile + (s & 1) * (BSTAGE_BYTES / 2);
        const int hcol = s * 16 + m;
        const float pv  = phb[hcol];
        const float wvv = wa[hcol];

        // Load ALL 7 B-fragments first (one big ds clause, latency overlapped),
        // then run the 7-WMMA accumulation chain.
        v16bf bfr[NKT];
        #pragma unroll
        for (int kt = 0; kt < NKT; ++kt)
            bfr[kt] = *(const v16bf*)(lb + ((kt * 32 + lane) << 4));

        v8f acc = {};
        #pragma unroll
        for (int kt = 0; kt < NKT; ++kt)
            acc = __builtin_amdgcn_wmma_f32_16x16x32_bf16(
                false, afrag[kt], false, bfr[kt], (short)0, acc, false, false);

        #pragma unroll
        for (int i = 0; i < 8; ++i)
            part[i] += dev_tanh(acc[i] + pv) * wvv;
    }

    // ---- reduce over the 16 D-columns per channel row ----
    // acc element i lives at (row = i + 8*hi16, col = m) of the 16x16 D tile.
    #pragma unroll
    for (int i = 0; i < 8; ++i)
        red[wv * 256 + (hi16 * 8 + i) * 16 + m] = part[i];
    __syncthreads();

    if (tid < 128) {
        int ww = tid >> 4, r = tid & 15;
        float sum = 0.f;
        #pragma unroll
        for (int n = 0; n < 16; ++n) sum += red[ww * 256 + r * 16 + n];
        scores[(size_t)b * CC + c0 + ww * 16 + r] = sum + ba[0];
    }
}

// ---------------------------------------------------------------------------
// Kernel 4: in-place softmax over C=2048 per batch row.
// ---------------------------------------------------------------------------
__global__ void k_softmax(float* __restrict__ wts) {
    __shared__ float buf[CC];
    __shared__ float rbuf[8];
    const int b = blockIdx.x, tid = threadIdx.x;
    float* p = wts + (size_t)b * CC;

    float mx = -3.4e38f;
    for (int i = tid; i < CC; i += 256) { float v = p[i]; buf[i] = v; mx = fmaxf(mx, v); }
    #pragma unroll
    for (int o = 16; o > 0; o >>= 1) mx = fmaxf(mx, __shfl_xor(mx, o, 32));
    if ((tid & 31) == 0) rbuf[tid >> 5] = mx;
    __syncthreads();
    mx = rbuf[0];
    #pragma unroll
    for (int i = 1; i < 8; ++i) mx = fmaxf(mx, rbuf[i]);
    __syncthreads();

    float sum = 0.f;
    for (int i = tid; i < CC; i += 256) { float e = __expf(buf[i] - mx); buf[i] = e; sum += e; }
    #pragma unroll
    for (int o = 16; o > 0; o >>= 1) sum += __shfl_xor(sum, o, 32);
    if ((tid & 31) == 0) rbuf[tid >> 5] = sum;
    __syncthreads();
    sum = 0.f;
    #pragma unroll
    for (int i = 0; i < 8; ++i) sum += rbuf[i];
    float inv = __frcp_rn(sum);
    for (int i = tid; i < CC; i += 256) p[i] = buf[i] * inv;
}

// ---------------------------------------------------------------------------
// Kernel 5: weighted[b,l] = sum_c weights[b,c] * att[b,l,c]
// One block per batch; wave per row; lanes coalesce over c with float4.
// ---------------------------------------------------------------------------
__global__ void k_weighted(const float* __restrict__ att, const float* __restrict__ wts,
                           float* __restrict__ out) {
    __shared__ float wsm[CC];
    const int b = blockIdx.x, tid = threadIdx.x;
    const int wv = tid >> 5, lane = tid & 31;
    for (int i = tid; i < CC; i += 256) wsm[i] = wts[(size_t)b * CC + i];
    __syncthreads();
    const float* abase = att + (size_t)b * LL * CC;
    for (int l = wv; l < LL; l += 8) {
        const float4* row = (const float4*)(abase + (size_t)l * CC);
        const float4* wrow = (const float4*)wsm;
        float s = 0.f;
        for (int c = lane; c < CC / 4; c += 32) {
            float4 r = row[c];
            float4 w4 = wrow[c];
            s = fmaf(r.x, w4.x, fmaf(r.y, w4.y, fmaf(r.z, w4.z, fmaf(r.w, w4.w, s))));
        }
        #pragma unroll
        for (int o = 16; o > 0; o >>= 1) s += __shfl_xor(s, o, 32);
        if (lane == 0) out[(size_t)b * LL + l] = s;
    }
}

// ---------------------------------------------------------------------------
// Launcher
// ---------------------------------------------------------------------------
extern "C" void kernel_launch(void* const* d_in, const int* in_sizes, int n_in,
                              void* d_out, int out_size, void* d_ws, size_t ws_size,
                              hipStream_t stream) {
    const float* att = (const float*)d_in[0];   // (128,196,2048)
    const float* h   = (const float*)d_in[1];   // (128,1024)
    const float* Wc  = (const float*)d_in[2];   // (196,512)
    const float* bc  = (const float*)d_in[3];   // (512,)
    const float* Wh  = (const float*)d_in[4];   // (1024,512)
    const float* bh  = (const float*)d_in[5];   // (512,)
    const float* wa  = (const float*)d_in[6];   // (512,)
    const float* ba  = (const float*)d_in[7];   // (1,)

    // workspace: [0, 229376) packed Wc bf16; [229376, 491520) ph fp32
    unsigned short* wcp = (unsigned short*)d_ws;
    float* ph = (float*)((char*)d_ws + NHT * NKT * 32 * 16 * sizeof(unsigned short));

    float* weighted = (float*)d_out;            // (128,196)
    float* weights  = (float*)d_out + BB * LL;  // (128,2048): scores -> softmax in place

    k_pack_wc<<<(NHT * NKT * 32 * 16 + 255) / 256, 256, 0, stream>>>(Wc, wcp);
    k_ph<<<(BB * HH + 255) / 256, 256, 0, stream>>>(h, Wh, bh, bc, ph);
    k_scores<<<dim3(CC / CTILE, BB), 256, 0, stream>>>(att, ph, wa, wcp, ba, weights);
    k_softmax<<<BB, 256, 0, stream>>>(weights);
    k_weighted<<<BB, 256, 0, stream>>>(att, weights, weighted);
}
